// StructuralAttentionLayer_23201413333109
// MI455X (gfx1250) — matrix operand order repaired and checked by reading, not compiled, not run on hardware
//
#include <hip/hip_runtime.h>
#include <math.h>

// Problem dims
#define BDIM 8
#define SDIM 512
#define HIDN 1024
#define HDIM 512
#define KDIM 37
#define BS   (BDIM * SDIM)    // 4096 rows
#define HK   (HDIM * KDIM)    // 18944

// xp GEMM tiling (f32 WMMA)
#define TM 64
#define TN 64
#define KC 64
#define LDT 66

// big GEMM tiling (split-bf16 WMMA)
#define TM2 128
#define TN2 128
#define PH  64                 // h-window per phase (2368 = 64*37 hk per phase)
#define PHK (PH * KDIM)        // 2368
#define LXP 68                 // xp window stride (floats)
#define LDB 72                 // bf16 LDS stride: keeps 16B fragment alignment

typedef __attribute__((ext_vector_type(2)))  float v2f;
typedef __attribute__((ext_vector_type(8)))  float v8f;
typedef __bf16 bf16;
typedef __attribute__((ext_vector_type(8)))  __bf16 v8bf;
typedef __attribute__((ext_vector_type(16))) __bf16 v16bf;

__device__ __forceinline__ v8f wmma4(v2f a, v2f b, v8f c) {
  return __builtin_amdgcn_wmma_f32_16x16x4_f32(false, a, false, b, (short)0, c, false, false);
}
__device__ __forceinline__ v8f wmma_bf(v16bf a, v16bf b, v8f c) {
  return __builtin_amdgcn_wmma_f32_16x16x32_bf16(false, a, false, b, (short)0, c, false, false);
}

__device__ __forceinline__ unsigned pk(bf16 a, bf16 b) {
  unsigned short ua = __builtin_bit_cast(unsigned short, a);
  unsigned short ub = __builtin_bit_cast(unsigned short, b);
  return (unsigned)ua | ((unsigned)ub << 16);
}

// A fragment (16x32 bf16): lane = row, K = 16h..16h+15 contiguous
__device__ __forceinline__ v16bf ldfragA(const bf16* rowp, int kk, int half) {
  return *(const v16bf*)(rowp + kk + 16 * half);
}
// B fragment (32x16 bf16): lane = column, K = 16h..16h+15 contiguous
__device__ __forceinline__ v16bf ldfragB(const bf16* rowp, int kk, int half) {
  return *(const v16bf*)(rowp + kk + 16 * half);
}

// async global->LDS 16B copy (GLOBAL_LOAD_ASYNC_TO_LDS_B128, ASYNCcnt-tracked)
__device__ __forceinline__ void async_cp16(unsigned lds_byte_addr,
                                           unsigned glb_byte_off,
                                           const float* __restrict__ sbase) {
  asm volatile("global_load_async_to_lds_b128 %0, %1, %2"
               :: "v"(lds_byte_addr), "v"(glb_byte_off), "s"(sbase) : "memory");
}
__device__ __forceinline__ void wait_async0() {
  asm volatile("s_wait_asynccnt 0x0" ::: "memory");
}

// ---------------------------------------------------------------------------
// scale[h*37+k] = sum_m head_WV[k,h,m]
// ---------------------------------------------------------------------------
__global__ void wsum_kernel(const float* __restrict__ head_WV,
                            float* __restrict__ scale_hk) {
  const int row = blockIdx.x;        // row = k*HDIM + h
  const int k = row >> 9;
  const int h = row & 511;
  const int t = threadIdx.x;         // 128 threads, 512 floats per row
  const float4 v = ((const float4*)(head_WV + (size_t)row * HDIM))[t];
  float s = v.x + v.y + v.z + v.w;
  #pragma unroll
  for (int off = 16; off; off >>= 1) s += __shfl_xor(s, off, 32);
  __shared__ float red[4];
  if ((t & 31) == 0) red[t >> 5] = s;
  __syncthreads();
  if (t == 0) scale_hk[h * KDIM + k] = red[0] + red[1] + red[2] + red[3];
}

// ---------------------------------------------------------------------------
// Asum[bi,k] = sum_j softmax_j(s_arc[bi,:]) * mask[bi] * softmax_k(s_rel[bi,j,:])
// ---------------------------------------------------------------------------
__global__ void asum_kernel(const float* __restrict__ s_arc,
                            const float* __restrict__ s_rel,
                            const float* __restrict__ mask,
                            float* __restrict__ Asum) {
  const int bi  = blockIdx.x;
  const int tid = threadIdx.x;
  __shared__ float pj[SDIM];
  __shared__ float red[8];
  __shared__ float asum_s[KDIM];

  const float* arow = s_arc + (size_t)bi * SDIM;
  const float a0 = arow[tid], a1 = arow[tid + 256];

  float m = fmaxf(a0, a1);
  #pragma unroll
  for (int off = 16; off; off >>= 1) m = fmaxf(m, __shfl_xor(m, off, 32));
  if ((tid & 31) == 0) red[tid >> 5] = m;
  __syncthreads();
  float rowmax = red[0];
  #pragma unroll
  for (int i = 1; i < 8; ++i) rowmax = fmaxf(rowmax, red[i]);
  __syncthreads();

  const float e0 = expf(a0 - rowmax), e1 = expf(a1 - rowmax);
  pj[tid] = e0; pj[tid + 256] = e1;
  float s = e0 + e1;
  #pragma unroll
  for (int off = 16; off; off >>= 1) s += __shfl_xor(s, off, 32);
  if ((tid & 31) == 0) red[tid >> 5] = s;
  if (tid < KDIM) asum_s[tid] = 0.0f;
  __syncthreads();
  float rowsum = 0.0f;
  #pragma unroll
  for (int i = 0; i < 8; ++i) rowsum += red[i];
  const float wscale = mask[bi] / rowsum;

  float acc[KDIM];
  #pragma unroll
  for (int k = 0; k < KDIM; ++k) acc[k] = 0.0f;

  for (int j = tid; j < SDIM; j += 256) {
    const float* rr = s_rel + ((size_t)bi * SDIM + j) * KDIM;
    float v[KDIM];
    float mx = -3.4e38f;
    #pragma unroll
    for (int k = 0; k < KDIM; ++k) { v[k] = rr[k]; mx = fmaxf(mx, v[k]); }
    float ssum = 0.0f;
    #pragma unroll
    for (int k = 0; k < KDIM; ++k) { v[k] = expf(v[k] - mx); ssum += v[k]; }
    const float w = pj[j] * wscale / ssum;
    #pragma unroll
    for (int k = 0; k < KDIM; ++k) acc[k] += v[k] * w;
  }

  #pragma unroll
  for (int k = 0; k < KDIM; ++k) {
    float x = acc[k];
    #pragma unroll
    for (int off = 16; off; off >>= 1) x += __shfl_xor(x, off, 32);
    if ((tid & 31) == 0) atomicAdd(&asum_s[k], x);
  }
  __syncthreads();
  if (tid < KDIM) Asum[(size_t)bi * KDIM + tid] = asum_s[tid];
}

// ---------------------------------------------------------------------------
// xp = x @ Wp + bp   (M=4096, N=512, K=1024), f32 WMMA, 64x64 tile
// ---------------------------------------------------------------------------
__global__ void __launch_bounds__(256)
xp_kernel(const float* __restrict__ x, const float* __restrict__ Wp,
          const float* __restrict__ bp, float* __restrict__ xp) {
  __shared__ float As[TM * LDT];
  __shared__ float Bs[TN * LDT];
  const int tid  = threadIdx.x;
  const int row0 = blockIdx.x * TM;
  const int col0 = blockIdx.y * TN;
  const int lane = tid & 31;
  const int wave = tid >> 5;
  const int wm   = wave & 3;
  const int wn0  = (wave >> 2) * 2;
  const int half = lane >> 4;
  const int lr   = lane & 15;

  v8f acc0 = {}; v8f acc1 = {};

  for (int k0 = 0; k0 < HIDN; k0 += KC) {
    __syncthreads();
    for (int idx = tid; idx < TM * KC; idx += 256) {
      const int r = idx >> 6, c = idx & 63;
      As[r * LDT + c] = x[(size_t)(row0 + r) * HIDN + k0 + c];
    }
    for (int idx = tid; idx < TN * KC; idx += 256) {
      const int c = idx >> 6, n = idx & 63;
      Bs[n * LDT + c] = Wp[(size_t)(k0 + c) * HDIM + col0 + n];
    }
    __syncthreads();
    #pragma unroll
    for (int kc = 0; kc < KC; kc += 4) {
      const v2f a  = *(const v2f*)&As[(wm * 16 + lr) * LDT + kc + 2 * half];
      const v2f b0 = *(const v2f*)&Bs[(wn0 * 16 + lr) * LDT + kc + 2 * half];
      const v2f b1 = *(const v2f*)&Bs[((wn0 + 1) * 16 + lr) * LDT + kc + 2 * half];
      acc0 = wmma4(a, b0, acc0);
      acc1 = wmma4(a, b1, acc1);
    }
  }
  #pragma unroll
  for (int v = 0; v < 8; ++v) {
    const int r  = row0 + wm * 16 + v + 8 * half;
    const int c  = col0 + wn0 * 16 + lr;
    xp[(size_t)r * HDIM + c]      = acc0[v] + bp[c];
    xp[(size_t)r * HDIM + c + 16] = acc1[v] + bp[c + 16];
  }
}

// ---------------------------------------------------------------------------
// out = gelu( flat @ Wd + bd ),  flat[bi,h*37+k] = xp[bi,h]*Asum[bi,k]*scale[hk]
// 128x128 tile, 8 h-phases of 64 (xp window async-copied to LDS), split-bf16
// WMMA (Ah*Bh + Ah*Bl + Al*Bh) with f32 accumulate. LDS ~134 KB -> 2 WG/WGP.
// ---------------------------------------------------------------------------
__global__ void __launch_bounds__(256)
big_kernel(const float* __restrict__ xp, const float* __restrict__ Asum,
           const float* __restrict__ scale_hk, const float* __restrict__ Wd,
           const float* __restrict__ bd, float* __restrict__ out) {
  extern __shared__ float smem[];
  float* xp_w   = smem;                      // 128*68 f32
  float* asum_s = xp_w + TM2 * LXP;          // 128*38 f32
  float* scl_w  = asum_s + TM2 * 38;         // 2368 f32
  bf16*  Ah     = (bf16*)(scl_w + PHK);      // 128*72 bf16 per plane
  bf16*  Al     = Ah + TM2 * LDB;
  bf16*  Bh     = Al + TM2 * LDB;
  bf16*  Bl     = Bh + TN2 * LDB;

  const int tid  = threadIdx.x;
  const int row0 = blockIdx.x * TM2;
  const int col0 = blockIdx.y * TN2;
  const int lane = tid & 31;
  const int wave = tid >> 5;
  const int wm   = wave & 3;       // 4 M-groups of 32 rows
  const int wn   = wave >> 2;      // 2 N-groups of 64 cols
  const int half = lane >> 4;
  const int lr   = lane & 15;

  // asum tile staged once (first chunk barrier publishes it)
  for (int idx = tid; idx < TM2 * KDIM; idx += 256) {
    const int r = idx / KDIM, k = idx - r * KDIM;
    asum_s[r * 38 + k] = Asum[(size_t)(row0 + r) * KDIM + k];
  }

  const int ar0 = (wm * 32 + lr) * LDB;
  const int ar1 = (wm * 32 + 16 + lr) * LDB;

  v8f acc[2][4];
  #pragma unroll
  for (int mi = 0; mi < 2; ++mi)
    #pragma unroll
    for (int ni = 0; ni < 4; ++ni) acc[mi][ni] = (v8f){};

  for (int h0 = 0; h0 < HDIM; h0 += PH) {
    __syncthreads();   // previous phase fully done with xp_w / scl_w
    // async global->LDS copy of the 128x64 xp window (16B per op)
    for (int idx = tid; idx < TM2 * (PH / 4); idx += 256) {
      const int r = idx >> 4, c4 = (idx & 15) * 4;
      const unsigned goff = (unsigned)(((row0 + r) * HDIM + h0 + c4) * 4);
      const unsigned loff = (unsigned)(size_t)(xp_w + r * LXP + c4);
      async_cp16(loff, goff, xp);
    }
    for (int idx = tid; idx < PHK; idx += 256)
      scl_w[idx] = scale_hk[h0 * KDIM + idx];
    wait_async0();     // own async loads complete; barrier below publishes

    for (int ck0 = 0; ck0 < PHK; ck0 += KC) {
      __syncthreads();
      // synthesize A chunk, split to bf16 hi/lo planes (2 elems -> b32 stores)
      for (int idx = tid; idx < TM2 * (KC / 2); idx += 256) {
        const int r = idx >> 5;
        const int c = (idx & 31) * 2;
        const unsigned l0 = (unsigned)(ck0 + c);
        const unsigned l1 = l0 + 1;
        const unsigned dh0 = l0 / 37u, dk0 = l0 - dh0 * 37u;
        const unsigned dh1 = l1 / 37u, dk1 = l1 - dh1 * 37u;
        const float v0 = xp_w[r * LXP + dh0] * asum_s[r * 38 + dk0] * scl_w[l0];
        const float v1 = xp_w[r * LXP + dh1] * asum_s[r * 38 + dk1] * scl_w[l1];
        const bf16 vh0 = (bf16)v0, vh1 = (bf16)v1;
        const bf16 vl0 = (bf16)(v0 - (float)vh0), vl1 = (bf16)(v1 - (float)vh1);
        *(unsigned*)(Ah + r * LDB + c) = pk(vh0, vh1);
        *(unsigned*)(Al + r * LDB + c) = pk(vl0, vl1);
      }
      // B chunk: Wd rows -> transposed bf16 hi/lo planes
      const int ghk0 = h0 * KDIM + ck0;
      for (int idx = tid; idx < (KC / 2) * TN2; idx += 256) {
        const int cp = idx >> 7;
        const int n  = idx & 127;
        const int c  = cp * 2;
        const float w0 = Wd[(size_t)(ghk0 + c) * HIDN + col0 + n];
        const float w1 = Wd[(size_t)(ghk0 + c + 1) * HIDN + col0 + n];
        const bf16 wh0 = (bf16)w0, wh1 = (bf16)w1;
        const bf16 wl0 = (bf16)(w0 - (float)wh0), wl1 = (bf16)(w1 - (float)wh1);
        *(unsigned*)(Bh + n * LDB + c) = pk(wh0, wh1);
        *(unsigned*)(Bl + n * LDB + c) = pk(wl0, wl1);
      }
      __syncthreads();
      #pragma unroll
      for (int kk = 0; kk < KC; kk += 32) {
        const v16bf ah0 = ldfragA(Ah + ar0, kk, half);
        const v16bf al0 = ldfragA(Al + ar0, kk, half);
        const v16bf ah1 = ldfragA(Ah + ar1, kk, half);
        const v16bf al1 = ldfragA(Al + ar1, kk, half);
        #pragma unroll
        for (int ni = 0; ni < 4; ++ni) {
          const int br = (wn * 64 + ni * 16 + lr) * LDB;
          const v16bf bh = ldfragB(Bh + br, kk, half);
          const v16bf bl = ldfragB(Bl + br, kk, half);
          acc[0][ni] = wmma_bf(ah0, bh, acc[0][ni]);
          acc[1][ni] = wmma_bf(ah1, bh, acc[1][ni]);
          acc[0][ni] = wmma_bf(ah0, bl, acc[0][ni]);
          acc[1][ni] = wmma_bf(ah1, bl, acc[1][ni]);
          acc[0][ni] = wmma_bf(al0, bh, acc[0][ni]);
          acc[1][ni] = wmma_bf(al1, bh, acc[1][ni]);
        }
      }
    }
  }

  // epilogue: + bd, exact GELU
  #pragma unroll
  for (int mi = 0; mi < 2; ++mi)
    #pragma unroll
    for (int ni = 0; ni < 4; ++ni)
      #pragma unroll
      for (int v = 0; v < 8; ++v) {
        const int r = row0 + wm * 32 + mi * 16 + v + 8 * half;
        const int c = col0 + wn * 64 + ni * 16 + lr;
        float y = acc[mi][ni][v] + bd[c];
        out[(size_t)r * HIDN + c] = 0.5f * y * (1.0f + erff(y * 0.70710678118654752f));
      }
}

// ---------------------------------------------------------------------------
extern "C" void kernel_launch(void* const* d_in, const int* in_sizes, int n_in,
                              void* d_out, int out_size, void* d_ws, size_t ws_size,
                              hipStream_t stream) {
  (void)in_sizes; (void)n_in; (void)out_size; (void)ws_size;
  const float* x       = (const float*)d_in[0];
  const float* s_arc   = (const float*)d_in[1];
  const float* s_rel   = (const float*)d_in[2];
  const float* mask    = (const float*)d_in[3];
  const float* Wp      = (const float*)d_in[4];
  const float* bp      = (const float*)d_in[5];
  const float* head_WV = (const float*)d_in[6];
  const float* Wd      = (const float*)d_in[7];
  const float* bd      = (const float*)d_in[8];
  float* out = (float*)d_out;

  float* xp_ws   = (float*)d_ws;                       // 4096*512 f32 (8 MB)
  float* Asum_ws = xp_ws + (size_t)BS * HDIM;          // 4096*37 f32
  float* scl_ws  = Asum_ws + (size_t)BS * KDIM;        // 18944 f32

  wsum_kernel<<<dim3(HK), dim3(128), 0, stream>>>(head_WV, scl_ws);
  asum_kernel<<<dim3(BS), dim3(256), 0, stream>>>(s_arc, s_rel, mask, Asum_ws);
  xp_kernel<<<dim3(BS / TM, HDIM / TN), dim3(256), 0, stream>>>(x, Wp, bp, xp_ws);

  const size_t smem = (size_t)(TM2 * LXP + TM2 * 38 + PHK) * sizeof(float)
                    + (size_t)(4 * 128 * LDB) * sizeof(bf16);   // 137,472 B
  big_kernel<<<dim3(BS / TM2, HIDN / TN2), dim3(256), smem, stream>>>(
      xp_ws, Asum_ws, scl_ws, Wd, bd, out);
}
